// MultiHeadAttention_5927054868593
// MI455X (gfx1250) — compile-verified, matrix operand order checked by reading, not run on hardware
//
#include <hip/hip_runtime.h>
#include <hip/hip_bf16.h>

// MHA forward for B=4, S=2048, D=1024, H=16, HS=64 on gfx1250 (CDNA5).
// GEMMs: v_wmma_f32_16x16x32_bf16 with Tensor Data Mover (tensor_load_to_lds)
// double-buffered global->LDS panel staging tracked by TENSORcnt.
// Attention: flash-style streaming softmax with TDM-fed K tiles and
// double-buffered V tiles; never materializes S x S.

typedef __attribute__((ext_vector_type(16))) __bf16 v16bf;
typedef __attribute__((ext_vector_type(8)))  __bf16 v8bf;
typedef __attribute__((ext_vector_type(8)))  float  v8f;
typedef __attribute__((ext_vector_type(4)))  unsigned int u32x4;
typedef __attribute__((ext_vector_type(8)))  unsigned int u32x8;

#define B_   4
#define S_   2048
#define D_   1024
#define H_   16
#define HS_  64

static __device__ __forceinline__ unsigned short f2bf(float f) {
    unsigned int u = __builtin_bit_cast(unsigned int, f);
    unsigned int r = (u + 0x7fffu + ((u >> 16) & 1u)) >> 16;
    return (unsigned short)r;
}

static __device__ __forceinline__ v16bf cat8(v8bf lo, v8bf hi) {
    return __builtin_shufflevector(lo, hi, 0,1,2,3,4,5,6,7,8,9,10,11,12,13,14,15);
}

static __device__ __forceinline__ v8f wmma_bf16(v16bf a, v16bf b, v8f c) {
    return __builtin_amdgcn_wmma_f32_16x16x32_bf16(
        /*neg_a=*/false, a, /*neg_b=*/false, b,
        /*c_mod=*/(short)0, c, /*reuse_a=*/false, /*reuse_b=*/false);
}

// ---------------------------------------------------------------------------
// Tensor Data Mover: DMA one TD1-row x TD0-col bf16 tile (row stride STRIDE
// elements) from global memory into LDS. Issued by a single wave; tracked by
// TENSORcnt. D# built in SGPRs per CDNA5 ISA 8.3/8.4 (2D tensor, groups 0+1).
// ---------------------------------------------------------------------------
template <unsigned TD0, unsigned TD1, unsigned STRIDE>
static __device__ __forceinline__ void tdm_load_tile(
    const unsigned short* gptr,   // wave-uniform: start of tile in global mem
    unsigned lds_off) {           // wave-uniform: LDS byte offset of dest
    unsigned long long ga = (unsigned long long)(__SIZE_TYPE__)gptr;
    u32x4 g0;
    g0[0] = 1u;                                   // count=1, user descriptor
    g0[1] = lds_off;                              // lds_addr[63:32]
    g0[2] = (unsigned)(ga & 0xffffffffu);         // global_addr[95:64]
    g0[3] = (unsigned)((ga >> 32) & 0x01ffffffu)  // global_addr[120:96]
          | (2u << 30);                           // type=2 ("image")
    u32x8 g1;
    g1[0] = 0x00010000u;                          // data_size=1 (2 bytes)
    g1[1] = (STRIDE & 0xffffu) << 16;             // tensor_dim0 lo16
    g1[2] = ((STRIDE >> 16) & 0xffffu)            // tensor_dim0 hi16
          | ((0x100000u & 0xffffu) << 16);        // tensor_dim1 lo16 (large)
    g1[3] = ((0x100000u >> 16) & 0xffffu)         // tensor_dim1 hi16
          | (TD0 << 16);                          // tile_dim0 elements
    g1[4] = TD1;                                  // tile_dim1 rows, tile_dim2=0
    g1[5] = STRIDE;                               // tensor_dim0_stride lo32
    g1[6] = 0u;                                   // stride hi / dim1_stride lo
    g1[7] = 0u;
    asm volatile("tensor_load_to_lds %0, %1" :: "s"(g0), "s"(g1) : "memory");
}

// ---------------------------------------------------------------------------
// fp32 -> bf16 conversion (RNE), 4 elements per thread.
// ---------------------------------------------------------------------------
__global__ void cvt_f32_to_bf16(const float* __restrict__ in,
                                unsigned short* __restrict__ out, int n) {
    int i = (blockIdx.x * blockDim.x + threadIdx.x) * 4;
    if (i + 3 < n) {
        float4 v = *(const float4*)(in + i);
        unsigned int p0 = (unsigned int)f2bf(v.x) | ((unsigned int)f2bf(v.y) << 16);
        unsigned int p1 = (unsigned int)f2bf(v.z) | ((unsigned int)f2bf(v.w) << 16);
        uint2 o; o.x = p0; o.y = p1;
        *(uint2*)(out + i) = o;
    } else {
        for (int j = i; j < n; ++j) out[j] = f2bf(in[j]);
    }
}

// ---------------------------------------------------------------------------
// GEMM core: C[64x64 tile] = A[M x 1024] * W^T  (W is [N=1024,K=1024] row-major,
// so the B operand column n is W row n -> contiguous K runs for both operands).
// Block: 256 threads = 8 waves, wave grid 4(M) x 2(N), each wave 16x32.
// K panels (64x32 bf16) DMA'd into double-buffered LDS by the TDM; wave 0
// issues next-panel loads while all waves run WMMA on the current panel.
// ---------------------------------------------------------------------------
struct GemmAcc { v8f c0, c1; };

static __device__ __forceinline__ GemmAcc gemm_tile_core(
    const unsigned short* __restrict__ A,   // [Mtotal, 1024] bf16
    const unsigned short* __restrict__ W,   // [1024, 1024]  bf16 (N,K)
    int mbase, int nbase,
    unsigned short (*As)[64][32], unsigned short (*Bs)[64][32]) {

    const int K = D_;
    const int NSTEP = K / 32;                 // 32
    int t = threadIdx.x;
    int wave = t >> 5, lane = t & 31;
    int wm = wave >> 1, wn = wave & 1;
    int lh = lane >> 4;                       // lane half
    int ll = lane & 15;
    int kb = lh * 8;

    const unsigned short* Arow = A + (size_t)mbase * K;
    const unsigned short* Wrow = W + (size_t)nbase * K;
    unsigned aLds = (unsigned)(__SIZE_TYPE__)&As[0][0][0];
    unsigned bLds = (unsigned)(__SIZE_TYPE__)&Bs[0][0][0];
    const unsigned bufBytes = 64 * 32 * 2;    // 4 KiB per panel

    v8f c0 = {0.f,0.f,0.f,0.f,0.f,0.f,0.f,0.f};
    v8f c1 = c0;

    if (wave == 0) {                          // preload panel 0
        tdm_load_tile<32, 64, 1024>(Arow, aLds);
        tdm_load_tile<32, 64, 1024>(Wrow, bLds);
    }

    for (int i = 0; i < NSTEP; ++i) {
        int buf = i & 1;
        if (wave == 0) {
            if (i + 1 < NSTEP) {              // kick off next panel, then wait
                int nbuf = buf ^ 1;           // for current one (in-order cnt)
                tdm_load_tile<32, 64, 1024>(Arow + (i + 1) * 32, aLds + nbuf * bufBytes);
                tdm_load_tile<32, 64, 1024>(Wrow + (i + 1) * 32, bLds + nbuf * bufBytes);
                __builtin_amdgcn_s_wait_tensorcnt(2);
            } else {
                __builtin_amdgcn_s_wait_tensorcnt(0);
            }
        }
        __syncthreads();                      // publish panel to all waves

        v8bf alo = *(const v8bf*)&As[buf][wm * 16 + ll][kb];
        v8bf ahi = *(const v8bf*)&As[buf][wm * 16 + ll][kb + 16];
        v16bf a = cat8(alo, ahi);

        v8bf b0lo = *(const v8bf*)&Bs[buf][wn * 32 + ll][kb];
        v8bf b0hi = *(const v8bf*)&Bs[buf][wn * 32 + ll][kb + 16];
        v16bf b0 = cat8(b0lo, b0hi);

        v8bf b1lo = *(const v8bf*)&Bs[buf][wn * 32 + 16 + ll][kb];
        v8bf b1hi = *(const v8bf*)&Bs[buf][wn * 32 + 16 + ll][kb + 16];
        v16bf b1 = cat8(b1lo, b1hi);

        c0 = wmma_bf16(a, b0, c0);
        c1 = wmma_bf16(a, b1, c1);

        __syncthreads();                      // all reads done before refill
    }
    GemmAcc r; r.c0 = c0; r.c1 = c1;
    return r;
}

// Q/K/V projection: writes bf16 output in head-major layout [B, H, S, HS].
__global__ __launch_bounds__(256) void gemm_qkv_kernel(
    const unsigned short* __restrict__ A,     // xb  [B*S, 1024]
    const unsigned short* __restrict__ W,     // Wb  [1024, 1024]
    unsigned short* __restrict__ Out) {       // [B, H, S, HS] bf16
    __shared__ __align__(16) unsigned short As[2][64][32];
    __shared__ __align__(16) unsigned short Bs[2][64][32];

    int mbase = blockIdx.x * 64;
    int nbase = blockIdx.y * 64;
    GemmAcc acc = gemm_tile_core(A, W, mbase, nbase, As, Bs);

    int lane = threadIdx.x & 31, wave = threadIdx.x >> 5;
    int wm = wave >> 1, wn = wave & 1, lh = lane >> 4, ll = lane & 15;
    for (int i = 0; i < 8; ++i) {
        int m = mbase + wm * 16 + lh * 8 + i;     // global row = b*S + s
        int b = m >> 11, s = m & (S_ - 1);
        int n0 = nbase + wn * 32 + ll;
        int n1 = n0 + 16;
        int h0 = n0 >> 6, hs0 = n0 & 63;
        int h1 = n1 >> 6, hs1 = n1 & 63;
        Out[(((size_t)(b * H_ + h0) * S_) + s) * HS_ + hs0] = f2bf(acc.c0[i]);
        Out[(((size_t)(b * H_ + h1) * S_) + s) * HS_ + hs1] = f2bf(acc.c1[i]);
    }
}

// Output projection: out = attn @ Wo^T + bo, fp32 stores.
__global__ __launch_bounds__(256) void gemm_out_kernel(
    const unsigned short* __restrict__ A,     // attn bf16 [B*S, 1024]
    const unsigned short* __restrict__ W,     // Wo bf16 [1024,1024]
    const float* __restrict__ bias,           // [1024]
    float* __restrict__ Out) {                // [B*S, 1024] fp32
    __shared__ __align__(16) unsigned short As[2][64][32];
    __shared__ __align__(16) unsigned short Bs[2][64][32];

    int mbase = blockIdx.x * 64;
    int nbase = blockIdx.y * 64;
    GemmAcc acc = gemm_tile_core(A, W, mbase, nbase, As, Bs);

    int lane = threadIdx.x & 31, wave = threadIdx.x >> 5;
    int wm = wave >> 1, wn = wave & 1, lh = lane >> 4, ll = lane & 15;
    for (int i = 0; i < 8; ++i) {
        int m = mbase + wm * 16 + lh * 8 + i;
        int n0 = nbase + wn * 32 + ll;
        int n1 = n0 + 16;
        Out[(size_t)m * D_ + n0] = acc.c0[i] + bias[n0];
        Out[(size_t)m * D_ + n1] = acc.c1[i] + bias[n1];
    }
}

// ---------------------------------------------------------------------------
// Flash attention (causal). One block = 128 query rows of one (b,h) head.
// 8 waves, each owning 16 query rows and the full 64-key tile.
// K tiles DMA'd by the TDM into double-buffered LDS (fetch kt+1 overlaps
// compute on kt); V tiles double-buffered with transposed manual staging.
// P re-distributed D->A layout via per-wave LDS scratch.
// ---------------------------------------------------------------------------
__global__ __launch_bounds__(256) void attn_kernel(
    const unsigned short* __restrict__ Qb,    // [B,H,S,HS] bf16
    const unsigned short* __restrict__ Kb,
    const unsigned short* __restrict__ Vb,
    unsigned short* __restrict__ Ob) {        // [B,S,D] bf16
    __shared__ __align__(16) unsigned short Ks[2][64][64];   // [key][hs]
    __shared__ __align__(16) unsigned short Vts[2][64][72];  // [hs][key], padded
    __shared__ __align__(16) unsigned short Ps[8][16][64];   // per-wave P scratch

    int bh = blockIdx.y;
    int b = bh >> 4, h = bh & (H_ - 1);
    int qbase = blockIdx.x * 128;

    int t = threadIdx.x;
    int wave = t >> 5, lane = t & 31;
    int lh = lane >> 4, ll = lane & 15;
    int kb = lh * 8;

    const size_t headoff = (size_t)bh * S_ * HS_;
    const unsigned short* Qh = Qb + headoff;
    const unsigned short* Kh = Kb + headoff;
    const unsigned short* Vh = Vb + headoff;

    unsigned kLds = (unsigned)(__SIZE_TYPE__)&Ks[0][0][0];
    const unsigned kBufBytes = 64 * 64 * 2;   // 8 KiB per K tile

    // Q fragments (A-operand, 16 rows x 64 K) held in registers for the block.
    int qrow = qbase + wave * 16 + ll;
    v16bf qf[2];
    for (int ks = 0; ks < 2; ++ks) {
        v8bf lo = *(const v8bf*)(Qh + (size_t)qrow * HS_ + ks * 32 + kb);
        v8bf hi = *(const v8bf*)(Qh + (size_t)qrow * HS_ + ks * 32 + kb + 16);
        qf[ks] = cat8(lo, hi);
    }

    v8f o[4];
    const v8f zero8 = {0.f,0.f,0.f,0.f,0.f,0.f,0.f,0.f};
    for (int nto = 0; nto < 4; ++nto) o[nto] = zero8;
    float mrow[8], lrow[8];
    for (int i = 0; i < 8; ++i) { mrow[i] = -3.0e38f; lrow[i] = 0.f; }

    const float scale = 0.125f;  // 1/sqrt(64)
    int ktmax = (qbase + 127) >> 6;

    // Staging indices for V: thread t covers key rows t/4..(two 8-wide chunks).
    int sRow = t >> 2;                 // key row 0..63
    int sCol = (t & 3) * 16;           // hs chunk base (two 8-chunks)

    // Preload tile 0: TDM for K, manual transposed staging for V.
    if (wave == 0) tdm_load_tile<64, 64, 64>(Kh, kLds);
    {
        #pragma unroll
        for (int c = 0; c < 2; ++c) {
            int col = sCol + c * 8;
            uint4 vv = *(const uint4*)(Vh + (size_t)sRow * HS_ + col);
            const unsigned short* pv = (const unsigned short*)&vv;
            #pragma unroll
            for (int j = 0; j < 8; ++j) Vts[0][col + j][sRow] = pv[j];
        }
    }

    for (int kt = 0; kt <= ktmax; ++kt) {
        int buf = kt & 1;
        int kbase = kt * 64;
        if (kt < ktmax) {
            // Stage tile kt+1 into the other buffer while kt computes.
            int nbuf = buf ^ 1;
            if (wave == 0)
                tdm_load_tile<64, 64, 64>(Kh + (size_t)(kbase + 64) * HS_,
                                          kLds + nbuf * kBufBytes);
            #pragma unroll
            for (int c = 0; c < 2; ++c) {
                int col = sCol + c * 8;
                uint4 vv = *(const uint4*)(Vh + (size_t)(kbase + 64 + sRow) * HS_ + col);
                const unsigned short* pv = (const unsigned short*)&vv;
                #pragma unroll
                for (int j = 0; j < 8; ++j) Vts[nbuf][col + j][sRow] = pv[j];
            }
            // Prefetch the tile after next on the V stream.
            if (kt + 2 <= ktmax)
                __builtin_prefetch(Vh + (size_t)(kbase + 128 + sRow) * HS_, 0, 0);
        }
        if (wave == 0) {
            if (kt < ktmax) __builtin_amdgcn_s_wait_tensorcnt(1);
            else            __builtin_amdgcn_s_wait_tensorcnt(0);
        }
        __syncthreads();                      // publish tile kt to all waves

        // S = Q K^T  (B-operand column n = key row n of Ks, contiguous K run).
        v8f sacc[4];
        for (int nt = 0; nt < 4; ++nt) sacc[nt] = zero8;
        #pragma unroll
        for (int nt = 0; nt < 4; ++nt) {
            #pragma unroll
            for (int ks = 0; ks < 2; ++ks) {
                v8bf blo = *(const v8bf*)&Ks[buf][nt * 16 + ll][ks * 32 + kb];
                v8bf bhi = *(const v8bf*)&Ks[buf][nt * 16 + ll][ks * 32 + kb + 16];
                sacc[nt] = wmma_bf16(qf[ks], cat8(blo, bhi), sacc[nt]);
            }
        }

        // Causal mask + online softmax. D-layout rows live across 16-lane halves.
        float p[4][8];
        #pragma unroll
        for (int i = 0; i < 8; ++i) {
            int qg = qbase + wave * 16 + lh * 8 + i;
            float rmax = -3.0e38f;
            #pragma unroll
            for (int nt = 0; nt < 4; ++nt) {
                int kg = kbase + nt * 16 + ll;
                float v = sacc[nt][i] * scale;
                if (kg > qg) v = -3.0e38f;
                p[nt][i] = v;
                rmax = fmaxf(rmax, v);
            }
            #pragma unroll
            for (int off = 1; off < 16; off <<= 1)
                rmax = fmaxf(rmax, __shfl_xor(rmax, off, 32));
            float mn = fmaxf(mrow[i], rmax);
            float alpha = __expf(mrow[i] - mn);
            mrow[i] = mn;
            float rsum = 0.f;
            #pragma unroll
            for (int nt = 0; nt < 4; ++nt) {
                float e = __expf(p[nt][i] - mn);
                p[nt][i] = e;
                rsum += e;
            }
            #pragma unroll
            for (int off = 1; off < 16; off <<= 1)
                rsum += __shfl_xor(rsum, off, 32);
            lrow[i] = lrow[i] * alpha + rsum;
            #pragma unroll
            for (int nto = 0; nto < 4; ++nto) o[nto][i] *= alpha;
        }

        // Re-distribute P from D-layout to A-layout through per-wave LDS.
        #pragma unroll
        for (int nt = 0; nt < 4; ++nt)
            #pragma unroll
            for (int i = 0; i < 8; ++i)
                Ps[wave][lh * 8 + i][nt * 16 + ll] = f2bf(p[nt][i]);

        v16bf pf[2];
        #pragma unroll
        for (int ks = 0; ks < 2; ++ks) {
            v8bf lo = *(const v8bf*)&Ps[wave][ll][ks * 32 + kb];
            v8bf hi = *(const v8bf*)&Ps[wave][ll][ks * 32 + kb + 16];
            pf[ks] = cat8(lo, hi);
        }

        // O += P V  (B-operand column n = hs row n of Vts, contiguous key run).
        #pragma unroll
        for (int nto = 0; nto < 4; ++nto) {
            #pragma unroll
            for (int ks = 0; ks < 2; ++ks) {
                v8bf vlo = *(const v8bf*)&Vts[buf][nto * 16 + ll][ks * 32 + kb];
                v8bf vhi = *(const v8bf*)&Vts[buf][nto * 16 + ll][ks * 32 + kb + 16];
                o[nto] = wmma_bf16(pf[ks], cat8(vlo, vhi), o[nto]);
            }
        }
        __syncthreads();                      // reads of buf done before refill
    }

    // Normalize and write bf16 output in [B, S, D] layout for the final GEMM.
    #pragma unroll
    for (int nto = 0; nto < 4; ++nto) {
        #pragma unroll
        for (int i = 0; i < 8; ++i) {
            int qg = qbase + wave * 16 + lh * 8 + i;
            int col = h * HS_ + nto * 16 + ll;
            float val = o[nto][i] / lrow[i];
            Ob[((size_t)(b * S_ + qg)) * D_ + col] = f2bf(val);
        }
    }
}

// ---------------------------------------------------------------------------
// Host launcher.
// ---------------------------------------------------------------------------
extern "C" void kernel_launch(void* const* d_in, const int* in_sizes, int n_in,
                              void* d_out, int out_size, void* d_ws, size_t ws_size,
                              hipStream_t stream) {
    (void)in_sizes; (void)n_in; (void)out_size; (void)ws_size;
    const float* x  = (const float*)d_in[0];
    const float* Wk = (const float*)d_in[1];
    const float* Wq = (const float*)d_in[2];
    const float* Wv = (const float*)d_in[3];
    const float* Wo = (const float*)d_in[4];
    const float* bo = (const float*)d_in[5];
    float* out = (float*)d_out;

    char* ws = (char*)d_ws;
    const size_t MB = 1024ull * 1024ull;
    unsigned short* xb   = (unsigned short*)(ws);             // 16 MiB
    unsigned short* wqb  = (unsigned short*)(ws + 16 * MB);   // 2 MiB each
    unsigned short* wkb  = (unsigned short*)(ws + 18 * MB);
    unsigned short* wvb  = (unsigned short*)(ws + 20 * MB);
    unsigned short* wob  = (unsigned short*)(ws + 22 * MB);
    unsigned short* Qb   = (unsigned short*)(ws + 24 * MB);   // 16 MiB each
    unsigned short* Kb   = (unsigned short*)(ws + 40 * MB);
    unsigned short* Vb   = (unsigned short*)(ws + 56 * MB);
    unsigned short* attb = (unsigned short*)(ws + 72 * MB);   // 16 MiB

    const int nx = B_ * S_ * D_;       // 8388608
    const int nw = D_ * D_;            // 1048576
    cvt_f32_to_bf16<<<(nx / 4 + 255) / 256, 256, 0, stream>>>(x,  xb,  nx);
    cvt_f32_to_bf16<<<(nw / 4 + 255) / 256, 256, 0, stream>>>(Wq, wqb, nw);
    cvt_f32_to_bf16<<<(nw / 4 + 255) / 256, 256, 0, stream>>>(Wk, wkb, nw);
    cvt_f32_to_bf16<<<(nw / 4 + 255) / 256, 256, 0, stream>>>(Wv, wvb, nw);
    cvt_f32_to_bf16<<<(nw / 4 + 255) / 256, 256, 0, stream>>>(Wo, wob, nw);

    dim3 gemm_grid((B_ * S_) / 64, D_ / 64);   // 128 x 16
    gemm_qkv_kernel<<<gemm_grid, 256, 0, stream>>>(xb, wqb, Qb);
    gemm_qkv_kernel<<<gemm_grid, 256, 0, stream>>>(xb, wkb, Kb);
    gemm_qkv_kernel<<<gemm_grid, 256, 0, stream>>>(xb, wvb, Vb);

    dim3 attn_grid(S_ / 128, B_ * H_);         // 16 x 64
    attn_kernel<<<attn_grid, 256, 0, stream>>>(Qb, Kb, Vb, attb);

    gemm_out_kernel<<<gemm_grid, 256, 0, stream>>>(attb, wob, bo, out);
}
